// AttentionSynapse_13280038879326
// MI455X (gfx1250) — matrix operand order repaired and checked by reading, not compiled, not run on hardware
//
#include <hip/hip_runtime.h>
#include <math.h>

// ---------------------------------------------------------------------------
// AttentionSynapse: out[b,q] = logsumexp_k( sum_e (gq@WQ')[q,e] * (gk@WK')[k,e] )
// with WQ' = WQ rows scaled by wmix[head].  bf16 WMMA pipeline for gfx1250,
// with GLOBAL_LOAD_ASYNC_TO_LDS_B128 staging of the shared K tile.
// ---------------------------------------------------------------------------

typedef __attribute__((ext_vector_type(8)))  __bf16 v8bf;
typedef __attribute__((ext_vector_type(16))) __bf16 v16bf;
typedef __attribute__((ext_vector_type(8)))  float  v8f;

#define WMMA_BF16(A, B, C) \
  __builtin_amdgcn_wmma_f32_16x16x32_bf16(false, (A), false, (B), (short)0, (C), false, false)

static constexpr int BATCH = 2;
static constexpr int TQ    = 2048;
static constexpr int TK    = 2048;
static constexpr int EDIM  = 1024;   // E == Eq == Ek
static constexpr int RQ    = BATCH * TQ;   // 4096 rows of Q' (batch-concat)
static constexpr int NKCHUNK = 8;          // Tk split across 8 workgroups

// ------------------------- async-LDS helpers -------------------------------
// gfx1250 async copy: global memory -> LDS at the vdst address, tracked by ASYNCcnt.
// LDS byte address = low 32 bits of the generic pointer (flat aperture rule).
__device__ inline void async_load_lds_b128(const void* gsrc, void* lds_dst) {
  const unsigned int laddr = (unsigned int)(unsigned long long)lds_dst;
  const unsigned long long gaddr = (unsigned long long)gsrc;
  asm volatile("global_load_async_to_lds_b128 %0, %1, off"
               :: "v"(laddr), "v"(gaddr) : "memory");
}
__device__ inline void wait_asynccnt0() {
  asm volatile("s_wait_asynccnt 0x0" ::: "memory");
}

// ------------------------- fragment loaders --------------------------------
// A fragment: 16(M) x 32(K) bf16, source row-major, stride ld (elements).
// Layout: lanes 0-15 hold M=lane, K 0..15; lanes 16-31 K 16..31 (matches B fragments).
__device__ inline v16bf load_frag_a(const __bf16* __restrict__ p, int ld) {
  const int lane = threadIdx.x & 31;
  const int m = lane & 15, hf = lane >> 4;
  const __bf16* base = p + (size_t)m * ld + hf * 16;
  v8bf lo = *(const v8bf*)(base);
  v8bf hi = *(const v8bf*)(base + 8);
  return __builtin_shufflevector(lo, hi, 0,1,2,3,4,5,6,7,8,9,10,11,12,13,14,15);
}

// B fragment: 32(K) x 16(N) bf16, taken from a row-major (N x K) source, i.e.
// B = src^T.  Contiguous-K layout (same as A): lanes 0-15 hold K=0..15 (N=lane), lanes 16-31 K=16..31.
// -> lane n reads 16 *contiguous* K values of row n: two b128 loads, no transpose.
__device__ inline v16bf load_frag_bt(const __bf16* __restrict__ p, int ld) {
  const int lane = threadIdx.x & 31;
  const int n = lane & 15, hf = lane >> 4;
  const __bf16* base = p + (size_t)n * ld + hf * 16;
  v8bf lo = *(const v8bf*)(base);
  v8bf hi = *(const v8bf*)(base + 8);
  return __builtin_shufflevector(lo, hi, 0,1,2,3,4,5,6,7,8,9,10,11,12,13,14,15);
}

// ------------------------- conversion kernels ------------------------------
__global__ __launch_bounds__(256) void cvt_bf16(const float* __restrict__ src,
                                                __bf16* __restrict__ dst, int n) {
  int i = blockIdx.x * blockDim.x + threadIdx.x;
  const int stride = gridDim.x * blockDim.x;
  for (; i < n; i += stride) dst[i] = (__bf16)src[i];
}

// WQ (E x Eq) row-major: scale row e by wmix[e/64] while converting.
__global__ __launch_bounds__(256) void cvt_wq(const float* __restrict__ src,
                                              const float* __restrict__ wmix,
                                              __bf16* __restrict__ dst, int n, int ld) {
  int i = blockIdx.x * blockDim.x + threadIdx.x;
  const int stride = gridDim.x * blockDim.x;
  for (; i < n; i += stride) {
    const int row = i / ld;
    dst[i] = (__bf16)(src[i] * wmix[row >> 6]);
  }
}

// ------------------------- projection GEMM ---------------------------------
// C[R x N] = A[R x K] @ W[N x K]^T, all row-major, bf16 in, bf16 out.
// WG tile 128M x 128N, 8 waves (4x2), wave tile 32M x 64N, k-step 32.
__global__ __launch_bounds__(256) void proj_gemm(const __bf16* __restrict__ A,
                                                 const __bf16* __restrict__ W,
                                                 __bf16* __restrict__ C,
                                                 int K, int N) {
  const int wave = threadIdx.x >> 5;
  const int wm = wave >> 1, wn = wave & 1;
  const int m0 = blockIdx.x * 128 + wm * 32;
  const int n0 = blockIdx.y * 128 + wn * 64;

  v8f acc[2][4] = {};
  for (int kk = 0; kk < K; kk += 32) {
    if (kk + 32 < K) {  // hint the next A tiles toward the caches
      __builtin_prefetch(A + (size_t)m0 * K + kk + 32, 0, 1);
      __builtin_prefetch(A + (size_t)(m0 + 16) * K + kk + 32, 0, 1);
    }
    v16bf a0 = load_frag_a(A + (size_t)m0 * K + kk, K);
    v16bf a1 = load_frag_a(A + (size_t)(m0 + 16) * K + kk, K);
#pragma unroll
    for (int j = 0; j < 4; ++j) {
      v16bf b = load_frag_bt(W + (size_t)(n0 + 16 * j) * K + kk, K);
      acc[0][j] = WMMA_BF16(a0, b, acc[0][j]);
      acc[1][j] = WMMA_BF16(a1, b, acc[1][j]);
    }
  }
  // C layout: VGPR r -> (M = r or r+8 by half-wave, N = lane&15)
  const int lane = threadIdx.x & 31;
  const int hf = lane >> 4, col = lane & 15;
#pragma unroll
  for (int i = 0; i < 2; ++i)
#pragma unroll
    for (int j = 0; j < 4; ++j)
#pragma unroll
      for (int r = 0; r < 8; ++r) {
        const int row = m0 + 16 * i + hf * 8 + r;
        C[(size_t)row * N + n0 + 16 * j + col] = (__bf16)acc[i][j][r];
      }
}

// ------------------------- fused scores + streaming LSE --------------------
// grid.x = 32 q-blocks (128 q rows), grid.y = 8 k-chunks (256 keys).
// Each wave: 16 q rows x 128 keys of f32 accumulators, e-loop in steps of 32
// with the 128x32 K-tile staged in LDS by GLOBAL_LOAD_ASYNC_TO_LDS_B128
// (double-buffered, shared by all 8 waves).
__global__ __launch_bounds__(256) void score_lse(const __bf16* __restrict__ Qp,
                                                 const __bf16* __restrict__ Kb,
                                                 float* __restrict__ pM,
                                                 float* __restrict__ pS) {
  // 80-byte padded rows: 40 bf16 per key row -> conflict-free half-wave ds reads
  __shared__ __align__(16) __bf16 kt[2][128 * 40];

  const int qblock = blockIdx.x;
  const int kchunk = blockIdx.y;
  const int wave = threadIdx.x >> 5;
  const int lane = threadIdx.x & 31;
  const int q0 = qblock * 128 + wave * 16;
  const int batch = qblock >> 4;                 // 16 q-blocks per batch
  const __bf16* Kbase = Kb + (size_t)batch * TK * EDIM;

  float m[8], s[8];
#pragma unroll
  for (int r = 0; r < 8; ++r) { m[r] = -INFINITY; s[r] = 0.f; }

  const int skey = threadIdx.x >> 1;             // staging: 2 threads per key row
  const int shf  = threadIdx.x & 1;

  for (int pass = 0; pass < 2; ++pass) {
    const int krow0 = kchunk * 256 + pass * 128;
    v8f acc[8] = {};

    {  // stage e-step 0 into buffer 0 (async, memory -> LDS directly)
      const __bf16* src = Kbase + (size_t)(krow0 + skey) * EDIM + shf * 16;
      __bf16* dst = &kt[0][skey * 40 + shf * 16];
      async_load_lds_b128(src, dst);
      async_load_lds_b128(src + 8, dst + 8);
    }

    for (int step = 0; step < 32; ++step) {
      // buffer staged last iteration (or prologue) must have landed in LDS:
      // each thread drains its own ASYNCcnt, then the WG barrier makes it global.
      wait_asynccnt0();
      __syncthreads();
      if (step + 1 < 32) {  // stage next buffer while computing current
        const __bf16* src =
            Kbase + (size_t)(krow0 + skey) * EDIM + (step + 1) * 32 + shf * 16;
        __bf16* dst = &kt[(step + 1) & 1][skey * 40 + shf * 16];
        async_load_lds_b128(src, dst);
        async_load_lds_b128(src + 8, dst + 8);
      }
      v16bf a = load_frag_a(Qp + (size_t)q0 * EDIM + step * 32, EDIM);
      const __bf16* tb = kt[step & 1];
      const int n = lane & 15, hf2 = lane >> 4;
#pragma unroll
      for (int t = 0; t < 8; ++t) {
        const __bf16* bb = tb + (t * 16 + n) * 40 + hf2 * 16;
        v8bf lo = ((const v8bf*)bb)[0], hi = ((const v8bf*)bb)[1];
        v16bf b = __builtin_shufflevector(lo, hi, 0,1,2,3,4,5,6,7,8,9,10,11,12,13,14,15);
        acc[t] = WMMA_BF16(a, b, acc[t]);
      }
    }

    // online logsumexp merge: per lane, VGPR r is one q-row, tiles t are 8 key cols
#pragma unroll
    for (int r = 0; r < 8; ++r) {
      float tmax = acc[0][r];
#pragma unroll
      for (int t = 1; t < 8; ++t) tmax = fmaxf(tmax, acc[t][r]);
      const float mn = fmaxf(m[r], tmax);
      float ssum = 0.f;
#pragma unroll
      for (int t = 0; t < 8; ++t) ssum += __expf(acc[t][r] - mn);
      s[r] = s[r] * __expf(m[r] - mn) + ssum;
      m[r] = mn;
    }
  }

  // reduce across the 16 lanes of each half-wave (the 16 key columns of a row)
#pragma unroll
  for (int r = 0; r < 8; ++r) {
    float mm = m[r], ss = s[r];
#pragma unroll
    for (int off = 8; off >= 1; off >>= 1) {
      float mo = __shfl_xor(mm, off, 16);
      float so = __shfl_xor(ss, off, 16);
      float mn = fmaxf(mm, mo);
      ss = ss * __expf(mm - mn) + so * __expf(mo - mn);
      mm = mn;
    }
    m[r] = mm; s[r] = ss;
  }

  if ((lane & 15) == 0) {
    const int hf = lane >> 4;
#pragma unroll
    for (int r = 0; r < 8; ++r) {
      const int row = q0 + hf * 8 + r;
      pM[(size_t)row * NKCHUNK + kchunk] = m[r];
      pS[(size_t)row * NKCHUNK + kchunk] = s[r];
    }
  }
}

// ------------------------- final reduce ------------------------------------
__global__ __launch_bounds__(256) void lse_finalize(const float* __restrict__ pM,
                                                    const float* __restrict__ pS,
                                                    float* __restrict__ out, int rows) {
  const int i = blockIdx.x * blockDim.x + threadIdx.x;
  if (i >= rows) return;
  float mm = -INFINITY;
#pragma unroll
  for (int c = 0; c < NKCHUNK; ++c) mm = fmaxf(mm, pM[(size_t)i * NKCHUNK + c]);
  float ss = 0.f;
#pragma unroll
  for (int c = 0; c < NKCHUNK; ++c)
    ss += pS[(size_t)i * NKCHUNK + c] * __expf(pM[(size_t)i * NKCHUNK + c] - mm);
  out[i] = mm + __logf(ss);   // BETA = 1
}

// ------------------------- host-side launch --------------------------------
extern "C" void kernel_launch(void* const* d_in, const int* in_sizes, int n_in,
                              void* d_out, int out_size, void* d_ws, size_t ws_size,
                              hipStream_t stream) {
  const float* gq = (const float*)d_in[0];   // (B,Tq,Eq)
  const float* gk = (const float*)d_in[1];   // (B,Tk,Ek)
  const float* WQ = (const float*)d_in[2];   // (E,Eq)
  const float* WK = (const float*)d_in[3];   // (E,Ek)
  const float* Wm = (const float*)d_in[4];   // (1,H)
  float* out = (float*)d_out;                // (B,1,Tq) = 4096 f32

  char* ws = (char*)d_ws;
  constexpr size_t SZ_G  = (size_t)RQ * EDIM * sizeof(__bf16);   // 8 MB
  constexpr size_t SZ_W  = (size_t)EDIM * EDIM * sizeof(__bf16); // 2 MB
  constexpr size_t SZ_P  = (size_t)RQ * NKCHUNK * sizeof(float); // 128 KB
  __bf16* gqb = (__bf16*)(ws);
  __bf16* gkb = (__bf16*)(ws + SZ_G);
  __bf16* wqb = (__bf16*)(ws + 2 * SZ_G);
  __bf16* wkb = (__bf16*)(ws + 2 * SZ_G + SZ_W);
  __bf16* Qp  = (__bf16*)(ws + 2 * SZ_G + 2 * SZ_W);
  __bf16* Kc  = (__bf16*)(ws + 3 * SZ_G + 2 * SZ_W);
  float*  pM  = (float*) (ws + 4 * SZ_G + 2 * SZ_W);
  float*  pS  = (float*) (ws + 4 * SZ_G + 2 * SZ_W + SZ_P);
  (void)in_sizes; (void)n_in; (void)out_size; (void)ws_size;

  // 1) convert inputs to bf16; fold wmix into WQ rows
  cvt_bf16<<<2048, 256, 0, stream>>>(gq, gqb, RQ * EDIM);
  cvt_bf16<<<2048, 256, 0, stream>>>(gk, gkb, RQ * EDIM);
  cvt_wq  <<<1024, 256, 0, stream>>>(WQ, Wm, wqb, EDIM * EDIM, EDIM);
  cvt_bf16<<<1024, 256, 0, stream>>>(WK, wkb, EDIM * EDIM);

  // 2) projections: Q' = gq @ WQ'^T, K = gk @ WK^T   (4096x1024x1024 each)
  proj_gemm<<<dim3(RQ / 128, EDIM / 128), 256, 0, stream>>>(gqb, wqb, Qp, EDIM, EDIM);
  proj_gemm<<<dim3(RQ / 128, EDIM / 128), 256, 0, stream>>>(gkb, wkb, Kc, EDIM, EDIM);

  // 3) fused scores + streaming logsumexp partials
  score_lse<<<dim3(RQ / 128, NKCHUNK), 256, 0, stream>>>(Qp, Kc, pM, pS);

  // 4) combine partials
  lse_finalize<<<(RQ + 255) / 256, 256, 0, stream>>>(pM, pS, out, RQ);
}
